// SpectralConvCross1d_51857435131961
// MI455X (gfx1250) — matrix operand order repaired and checked by reading, not compiled, not run on hardware
//
#include <hip/hip_runtime.h>
#include <math.h>

// ---------------------------------------------------------------------------
// SpectralConvCross1d on MI455X (gfx1250, wave32, WMMA bf16 -> f32 accum)
//
//   K0a/K0b : build DFT / inverse-DFT basis tables PRE-SWIZZLED into the
//             WMMA B-fragment lane layout (bf16) -> B frag = 2x b128 loads
//   K1      : Qf,Kf = truncated rfft of q,k ; A panels staged to LDS with
//             global_load_async_to_lds_b128 (double buffered, s_wait_asynccnt)
//   K2      : per (b,h): QK = Qf^T Kf (complex WMMA) -> complex tanh ->
//             XV = tanh(QK)·Kf (complex WMMA), QK held in LDS
//   K3      : S[b,o] = sum_e XV[b,e]*W[e,o] per (h,mode) (complex WMMA)
//   K4      : out = S x invBasis (WMMA, pre-swizzled bf16 basis)
// ---------------------------------------------------------------------------

typedef __bf16 bf16_t;
typedef bf16_t v16bf __attribute__((ext_vector_type(16)));
typedef float  v8f   __attribute__((ext_vector_type(8)));

#define B_   8
#define H_   8
#define E_   64
#define L_   2048
#define M_   64          // kept modes
#define C2_  128         // 2*M_ (re/im interleaved)
#define NROW 4096        // B*H*E rows per DFT matrix

// ---- WMMA fragment helpers (A/B in contiguous k order, wave32) --------------------

template <typename F>
__device__ __forceinline__ v16bf load_a_frag(int kbase, F&& at) {
  // A is 16x32 (MxK). lane l<16: M=l, K = kb+0..15 ; lane l>=16: M=l-16, K = kb+16..31
  const int lane = threadIdx.x & 31;
  const int m    = lane & 15;
  const int ksel = (lane >> 4) * 16;
  v16bf a;
#pragma unroll
  for (int j = 0; j < 16; ++j) {
    a[j] = (bf16_t)at(m, kbase + ksel + j);
  }
  return a;
}

template <typename F>
__device__ __forceinline__ v16bf load_b_frag(int kbase, F&& at) {
  // B is 32x16 (KxN). lane l<16: N=l, K = kb+0..15 ; lane l>=16: N=l-16, K = kb+16..31
  const int lane = threadIdx.x & 31;
  const int n    = lane & 15;
  const int ksel = (lane >> 4) * 16;
  v16bf b;
#pragma unroll
  for (int j = 0; j < 16; ++j) {
    b[j] = (bf16_t)at(kbase + ksel + j, n);
  }
  return b;
}

// pre-swizzled B fragment: one 32-byte (v16bf) load per lane
__device__ __forceinline__ v16bf load_b_pre(const bf16_t* __restrict__ base, int tile) {
  const int lane = threadIdx.x & 31;
  return *(const v16bf*)(base + ((size_t)tile * 32 + lane) * 16);
}

__device__ __forceinline__ v8f wmma_bf16(v16bf a, v16bf b, v8f c) {
  return __builtin_amdgcn_wmma_f32_16x16x32_bf16(
      /*neg_a=*/false, a, /*neg_b=*/false, b,
      /*c_mod=*/(short)0, c, /*reuse_a=*/false, /*reuse_b=*/false);
}

template <typename F>
__device__ __forceinline__ void store_c(v8f c, F&& st) {
  // C/D: lane l: N=l&15, VGPR j holds row M = j + 8*(l>>4)
  const int lane = threadIdx.x & 31;
  const int n    = lane & 15;
  const int mh   = (lane >> 4) * 8;
#pragma unroll
  for (int j = 0; j < 8; ++j) st(mh + j, n, c[j]);
}

// ---- K0: pre-swizzled basis tables (bf16, B-fragment layout) ---------------
// fwd layout: [kt(64)][ct(8)][lane(32)][j(16)] ; element (lane,j) holds
//   t = kt*32 + (lane>>4)*16 + j , c = ct*16 + (lane&15)

__global__ void k0_basis_fwd(bf16_t* __restrict__ bFs) {
  int i = blockIdx.x * blockDim.x + threadIdx.x;   // over 64*8*32*16
  if (i >= 64 * 8 * 32 * 16) return;
  int j = i & 15, lane = (i >> 4) & 31, ct = (i >> 9) & 7, kt = i >> 12;
  int t = kt * 32 + (lane >> 4) * 16 + j;          // 0..2047
  int c = ct * 16 + (lane & 15);                   // 0..127
  int m = c >> 1;
  int p = (m * t) & (L_ - 1);                      // exact mod-L periodicity
  float ang = (float)(6.283185307179586 / (double)L_) * (float)p;
  float v = (c & 1) ? -__sinf(ang) : __cosf(ang);
  bFs[i] = (bf16_t)v;
}

// inv layout: [kt(4)][ct(128)][lane(32)][j(16)] ; element holds
//   k = kt*32 + (lane>>4)*16 + j (0..127) , t = ct*16 + (lane&15)
__global__ void k0_basis_inv(bf16_t* __restrict__ bIs) {
  int i = blockIdx.x * blockDim.x + threadIdx.x;   // over 4*128*32*16
  if (i >= 4 * 128 * 32 * 16) return;
  int j = i & 15, lane = (i >> 4) & 31, ct = (i >> 9) & 127, kt = i >> 16;
  int k = kt * 32 + (lane >> 4) * 16 + j;          // 0..127
  int t = ct * 16 + (lane & 15);                   // 0..2047
  int m = k >> 1;
  int p = (m * t) & (L_ - 1);
  float ang = (float)(6.283185307179586 / (double)L_) * (float)p;
  // irfft of 64-mode half spectrum: (1/L)(Re X0 + 2*sum_{m>=1}(Re cos - Im sin))
  // fold in the /IN_CH/OUT_CH = 1/(512*512) scale; Im of bin 0 ignored.
  float coeff = ((m == 0) ? 1.0f : 2.0f) * (1.0f / (float)L_) * (1.0f / 262144.0f);
  float v;
  if (k & 1) v = (m == 0) ? 0.0f : -coeff * __sinf(ang);
  else       v = coeff * __cosf(ang);
  bIs[i] = (bf16_t)v;
}

// ---- K1: truncated forward DFT of q and k (stacked), async-LDS staged ------
// QKf[g][c], g in [0,8192): g<4096 -> q row, else k row; c in [0,128)
// One wave per 16-row tile; the 16 rows are 16 consecutive e of one (b,h),
// so each timestep t contributes 16 contiguous floats (64B) of A data.

#define KC 64   // timesteps staged per chunk (4 KB)

__device__ __forceinline__ void stage_chunk(const float* __restrict__ gbase,
                                            float* __restrict__ sbuf, int t0) {
  // 64 t x 16 floats; 8 async b128: lane -> (tloc = i*8 + lane/4, 16B unit = lane&3)
  const int lane = threadIdx.x & 31;
  const int tl   = lane >> 2;
  const int un   = lane & 3;
#pragma unroll
  for (int i = 0; i < 8; ++i) {
    int tloc = i * 8 + tl;
    const void* g = (const char*)(gbase + (size_t)(t0 + tloc) * 512) + un * 16;
    unsigned lds  = (unsigned)(uintptr_t)(sbuf + tloc * 16 + un * 4);
    asm volatile("global_load_async_to_lds_b128 %0, %1, off"
                 :: "v"(lds), "v"(g) : "memory");
  }
}

__global__ __launch_bounds__(32) void k1_fwd_dft(const float* __restrict__ q,
                                                 const float* __restrict__ k,
                                                 const bf16_t* __restrict__ bFs,
                                                 float* __restrict__ QKf) {
  __shared__ float sA[2][KC * 16];                 // ping-pong staging, 8 KB
  const int rt = blockIdx.x;                       // 512 row tiles of 16
  const int g0 = rt * 16;
  const float* src = (g0 < NROW) ? q : k;
  const int r0 = g0 & (NROW - 1);
  const int b0 = r0 >> 9, h0 = (r0 >> 6) & 7, e0 = r0 & 63;
  const float* gbase = src + (size_t)b0 * L_ * 512 + h0 * 64 + e0; // [t*512 + m]

  v8f acc[8];
#pragma unroll
  for (int ct = 0; ct < 8; ++ct) acc[ct] = (v8f){0.f,0.f,0.f,0.f,0.f,0.f,0.f,0.f};

  stage_chunk(gbase, sA[0], 0);
  for (int chunk = 0; chunk < L_ / KC; ++chunk) {
    if (chunk + 1 < L_ / KC) {
      stage_chunk(gbase, sA[(chunk + 1) & 1], (chunk + 1) * KC);
      asm volatile("s_wait_asynccnt 8" ::: "memory");   // chunk's 8 loads done
    } else {
      asm volatile("s_wait_asynccnt 0" ::: "memory");
    }
    const float* sb = sA[chunk & 1];
#pragma unroll
    for (int kt = 0; kt < KC / 32; ++kt) {
      v16bf a = load_a_frag(kt * 32, [&](int m, int kk) -> float {
        return sb[kk * 16 + m];
      });
      const int ktile_abs = (chunk * KC + kt * 32) >> 5;
#pragma unroll
      for (int ct = 0; ct < 8; ++ct) {
        v16bf bb = load_b_pre(bFs, ktile_abs * 8 + ct);
        acc[ct] = wmma_bf16(a, bb, acc[ct]);
      }
    }
  }
#pragma unroll
  for (int ct = 0; ct < 8; ++ct) {
    store_c(acc[ct], [&](int r, int n, float v) {
      QKf[(size_t)(g0 + r) * C2_ + ct * 16 + n] = v;
    });
  }
}

// ---- K2: per-(b,h) frequency-domain cross attention ------------------------

__device__ __forceinline__ void ctanh_dev(float a, float b, float& ro, float& io) {
  // tanh(a+ib) = (sinh 2a + i sin 2b) / (cosh 2a + cos 2b), clamped for safety
  float a2 = fminf(fmaxf(2.f * a, -30.f), 30.f);
  float b2 = 2.f * b;
  float sh = sinhf(a2), ch = coshf(a2);
  float sn = __sinf(b2), cs = __cosf(b2);
  float d = ch + cs;
  d = (fabsf(d) < 1e-30f) ? 1e-30f : d;
  ro = sh / d;
  io = sn / d;
}

__global__ __launch_bounds__(256) void k2_attn(const float* __restrict__ QKf,
                                               float* __restrict__ XV) {
  __shared__ float sRe[64 * 64];
  __shared__ float sIm[64 * 64];
  const int bh   = blockIdx.x;                     // 64 blocks = (b,h)
  const int wave = threadIdx.x >> 5;
  const float* Qf = QKf + (size_t)bh * 64 * C2_;          // rows e, cols 2m(+1)
  const float* Kf = QKf + (size_t)(NROW + bh * 64) * C2_;

  // ---- step 3: QK[x,y] = sum_e Qf[e,x] * Kf[e,y]   (complex) ----
  for (int tt = wave; tt < 16; tt += 8) {
    int tx = tt >> 2, ty = tt & 3;
    v8f cre = (v8f){0.f,0.f,0.f,0.f,0.f,0.f,0.f,0.f};
    v8f cim = cre;
    for (int kb = 0; kb < E_; kb += 32) {
      v16bf aqr = load_a_frag(kb, [&](int m, int kk) { return  Qf[(size_t)kk * C2_ + 2 * (tx * 16 + m)]; });
      v16bf aqi = load_a_frag(kb, [&](int m, int kk) { return  Qf[(size_t)kk * C2_ + 2 * (tx * 16 + m) + 1]; });
      v16bf aqn = load_a_frag(kb, [&](int m, int kk) { return -Qf[(size_t)kk * C2_ + 2 * (tx * 16 + m) + 1]; });
      v16bf bkr = load_b_frag(kb, [&](int kk, int n) { return  Kf[(size_t)kk * C2_ + 2 * (ty * 16 + n)]; });
      v16bf bki = load_b_frag(kb, [&](int kk, int n) { return  Kf[(size_t)kk * C2_ + 2 * (ty * 16 + n) + 1]; });
      cre = wmma_bf16(aqr, bkr, cre);
      cre = wmma_bf16(aqn, bki, cre);
      cim = wmma_bf16(aqr, bki, cim);
      cim = wmma_bf16(aqi, bkr, cim);
    }
    store_c(cre, [&](int r, int n, float v) { sRe[(tx * 16 + r) * 64 + ty * 16 + n] = v; });
    store_c(cim, [&](int r, int n, float v) { sIm[(tx * 16 + r) * 64 + ty * 16 + n] = v; });
  }
  __syncthreads();

  // ---- complex tanh in place ----
  for (int i = threadIdx.x; i < 64 * 64; i += 256) {
    float ro, io;
    ctanh_dev(sRe[i], sIm[i], ro, io);
    sRe[i] = ro;
    sIm[i] = io;
  }
  __syncthreads();

  // ---- step 5: XV[e,x] = sum_y QK[x,y] * Kf[e,y]   (complex) ----
  float* XVb = XV + (size_t)bh * 64 * C2_;
  for (int tt = wave; tt < 16; tt += 8) {
    int te = tt >> 2, tx = tt & 3;
    v8f xre = (v8f){0.f,0.f,0.f,0.f,0.f,0.f,0.f,0.f};
    v8f xim = xre;
    for (int kb = 0; kb < M_; kb += 32) {
      v16bf akr = load_a_frag(kb, [&](int m, int kk) { return  Kf[(size_t)(te * 16 + m) * C2_ + 2 * kk]; });
      v16bf aki = load_a_frag(kb, [&](int m, int kk) { return  Kf[(size_t)(te * 16 + m) * C2_ + 2 * kk + 1]; });
      v16bf akn = load_a_frag(kb, [&](int m, int kk) { return -Kf[(size_t)(te * 16 + m) * C2_ + 2 * kk + 1]; });
      v16bf bre = load_b_frag(kb, [&](int kk, int n) { return sRe[(tx * 16 + n) * 64 + kk]; });
      v16bf bim = load_b_frag(kb, [&](int kk, int n) { return sIm[(tx * 16 + n) * 64 + kk]; });
      xre = wmma_bf16(akr, bre, xre);
      xre = wmma_bf16(akn, bim, xre);
      xim = wmma_bf16(akr, bim, xim);
      xim = wmma_bf16(aki, bre, xim);
    }
    store_c(xre, [&](int r, int n, float v) { XVb[(size_t)(te * 16 + r) * C2_ + 2 * (tx * 16 + n)] = v; });
    store_c(xim, [&](int r, int n, float v) { XVb[(size_t)(te * 16 + r) * C2_ + 2 * (tx * 16 + n) + 1] = v; });
  }
}

// ---- K3: S[b,h,o,x] = sum_e XV[b,h,e,x] * W[h,e,o,x]  (complex) ------------
// one wave per (h, x, o-tile); batch rows b padded 8 -> 16

__global__ __launch_bounds__(32) void k3_mix(const float* __restrict__ XV,
                                             const float* __restrict__ W, // complex interleaved
                                             float* __restrict__ S) {
  const int bid = blockIdx.x;                      // 8*64*4 = 2048
  const int nt  = bid & 3;
  const int x   = (bid >> 2) & 63;
  const int h   = bid >> 8;

  v8f cre = (v8f){0.f,0.f,0.f,0.f,0.f,0.f,0.f,0.f};
  v8f cim = cre;
  for (int kb = 0; kb < E_; kb += 32) {
    v16bf are = load_a_frag(kb, [&](int m, int kk) {
      return (m < 8) ? XV[(size_t)((m * 8 + h) * 64 + kk) * C2_ + 2 * x] : 0.f;
    });
    v16bf aim = load_a_frag(kb, [&](int m, int kk) {
      return (m < 8) ? XV[(size_t)((m * 8 + h) * 64 + kk) * C2_ + 2 * x + 1] : 0.f;
    });
    v16bf ain = load_a_frag(kb, [&](int m, int kk) {
      return (m < 8) ? -XV[(size_t)((m * 8 + h) * 64 + kk) * C2_ + 2 * x + 1] : 0.f;
    });
    v16bf bre = load_b_frag(kb, [&](int kk, int n) {
      return W[2 * ((size_t)((h * 64 + kk) * 64 + (nt * 16 + n)) * 64 + x)];
    });
    v16bf bim = load_b_frag(kb, [&](int kk, int n) {
      return W[2 * ((size_t)((h * 64 + kk) * 64 + (nt * 16 + n)) * 64 + x) + 1];
    });
    cre = wmma_bf16(are, bre, cre);
    cre = wmma_bf16(ain, bim, cre);
    cim = wmma_bf16(are, bim, cim);
    cim = wmma_bf16(aim, bre, cim);
  }
  store_c(cre, [&](int r, int n, float v) {
    if (r < 8) S[(size_t)((r * 8 + h) * 64 + (nt * 16 + n)) * C2_ + 2 * x] = v;
  });
  store_c(cim, [&](int r, int n, float v) {
    if (r < 8) S[(size_t)((r * 8 + h) * 64 + (nt * 16 + n)) * C2_ + 2 * x + 1] = v;
  });
}

// ---- K4: inverse DFT: out[r,t] = sum_k S[r,k] * bI[k,t] --------------------

__global__ __launch_bounds__(32) void k4_inv_dft(const float* __restrict__ S,
                                                 const bf16_t* __restrict__ bIs,
                                                 float* __restrict__ out) {
  const int bid = blockIdx.x;                      // 256 row-tiles * 32 col-groups
  const int rt  = bid >> 5;
  const int cg  = bid & 31;                        // 64 cols each

  v8f acc[4];
#pragma unroll
  for (int ct = 0; ct < 4; ++ct) acc[ct] = (v8f){0.f,0.f,0.f,0.f,0.f,0.f,0.f,0.f};

  for (int kb = 0; kb < C2_; kb += 32) {
    v16bf a = load_a_frag(kb, [&](int m, int kk) {
      return S[(size_t)(rt * 16 + m) * C2_ + kk];
    });
    const int kt = kb >> 5;
#pragma unroll
    for (int ct = 0; ct < 4; ++ct) {
      v16bf bb = load_b_pre(bIs, kt * 128 + cg * 4 + ct);
      acc[ct] = wmma_bf16(a, bb, acc[ct]);
    }
  }
#pragma unroll
  for (int ct = 0; ct < 4; ++ct) {
    store_c(acc[ct], [&](int r, int n, float v) {
      out[(size_t)(rt * 16 + r) * L_ + cg * 64 + ct * 16 + n] = v;
    });
  }
}

// ---------------------------------------------------------------------------

extern "C" void kernel_launch(void* const* d_in, const int* in_sizes, int n_in,
                              void* d_out, int out_size, void* d_ws, size_t ws_size,
                              hipStream_t stream) {
  const float* q = (const float*)d_in[0];
  const float* k = (const float*)d_in[1];
  // d_in[2] = v : unused by the reference math
  const float* w = (const float*)d_in[3];  // complex64 -> interleaved float pairs
  float* out = (float*)d_out;

  // workspace layout:
  bf16_t* bFs = (bf16_t*)d_ws;                     // 262144 bf16 (512 KB), swizzled fwd basis
  bf16_t* bIs = bFs + 64 * 8 * 32 * 16;            // 262144 bf16 (512 KB), swizzled inv basis
  float*  QKf = (float*)(bIs + 4 * 128 * 32 * 16); // [8192][128] f32: Qf rows 0..4095, Kf 4096..8191
  float*  XV  = QKf + (size_t)2 * NROW * C2_;      // [4096][128] f32
  float*  S   = XV + (size_t)NROW * C2_;           // [4096][128] f32

  k0_basis_fwd<<<(64 * 8 * 32 * 16 + 255) / 256, 256, 0, stream>>>(bFs);
  k0_basis_inv<<<(4 * 128 * 32 * 16 + 255) / 256, 256, 0, stream>>>(bIs);
  k1_fwd_dft<<<512, 32, 0, stream>>>(q, k, bFs, QKf);
  k2_attn<<<64, 256, 0, stream>>>(QKf, XV);
  k3_mix<<<2048, 32, 0, stream>>>(XV, w, S);
  k4_inv_dft<<<8192, 32, 0, stream>>>(S, bIs, out);
}
